// Transformer_48378511622205
// MI455X (gfx1250) — matrix operand order repaired and checked by reading, not compiled, not run on hardware
//
#include <hip/hip_runtime.h>
#include <math.h>

// ---------------------------------------------------------------------------
// Transformer rollout as exact KV-cache autoregressive decoding.
// Single persistent workgroup, 512 threads = 16 wave32 (deep MLP for L2
// weight streaming); matvecs via v_wmma_f32_16x16x32_bf16 with bf16
// activations in LDS and biases folded into the WMMA C operand.
// ---------------------------------------------------------------------------

#define D      256
#define HD     32
#define NH     8
#define S      128
#define FF     1024
#define NL     3
#define EPSLN  1e-5f
#define TPB    512
#define NW     16          // waves per block

typedef __attribute__((ext_vector_type(16))) __bf16 v16bf;
typedef __attribute__((ext_vector_type(8)))  __bf16 v8bf;
typedef __attribute__((ext_vector_type(8)))  float  v8f;

__device__ __forceinline__ unsigned short f2bf_bits(float f) {
    union { float f; unsigned u; } v; v.f = f;
    unsigned u = v.u;
    unsigned r = u + 0x7fffu + ((u >> 16) & 1u);   // round-to-nearest-even
    return (unsigned short)(r >> 16);
}

// ---------------------------------------------------------------------------
// Pack scale*W[N,K] (row-major, y = W x) into WMMA B-fragment layout for
// v_wmma_f32_16x16x32_bf16: tile (nt,kt), lane = (n%16) + 16*(k%32>=16),
// element j = k%16, lane stride 16 elems (32 B contiguous per lane).
// ---------------------------------------------------------------------------
__global__ void pack_bf16_wmma(const float* __restrict__ W,
                               unsigned short* __restrict__ dst,
                               int N, int K, float scale) {
    int total = N * K;
    int Ktiles = K >> 5;
    for (int idx = blockIdx.x * blockDim.x + threadIdx.x; idx < total;
         idx += gridDim.x * blockDim.x) {
        int n  = idx / K, k = idx % K;
        int nt = n >> 4,  lr = n & 15;
        int kt = k >> 5,  ko = k & 31;
        int half = (ko >> 4) & 1;
        int lane = lr + (half << 4);
        int j    = ko & 15;
        size_t off = (((size_t)nt * Ktiles + kt) * 32 + lane) * 16 + j;
        dst[off] = f2bf_bits(W[idx] * scale);
    }
}

__global__ void pe_kernel(float* __restrict__ pe) {
    int idx = blockIdx.x * blockDim.x + threadIdx.x;
    if (idx < S * D) {
        int s = idx / D, d = idx % D;
        int pair = d >> 1;
        float div = __expf(-(float)(2 * pair) * (logf(10000.0f) / (float)D));
        float ang = (float)s * div;
        pe[idx] = (d & 1) ? cosf(ang) : sinf(ang);
    }
}

__device__ __forceinline__ v8f wmma_bf16(v16bf a, v16bf b, v8f c) {
    return __builtin_amdgcn_wmma_f32_16x16x32_bf16(false, a, false, b,
                                                   (short)0, c, false, false);
}

// Build A fragment (x broadcast over M) for this lane from bf16 LDS vector.
__device__ __forceinline__ v16bf a_frag(const __bf16* __restrict__ xb,
                                        int kt, int abase) {
    v8bf lo = *(const v8bf*)(xb + kt * 32 + abase);        // K = base+0..7
    v8bf hi = *(const v8bf*)(xb + kt * 32 + abase + 8);    // K = base+8..15
    return __builtin_shufflevector(lo, hi, 0, 1, 2, 3, 4, 5, 6, 7,
                                   8, 9, 10, 11, 12, 13, 14, 15);
}

// ---------------------------------------------------------------------------
// WMMA matvec with fused bias: y[0..N) = Wp * x[0..K) + bscale*bias.
// Bias enters through the WMMA C operand (C[m,n] = bias[n] for all m).
// A broadcast over rows => every row of D equals y; lane<16 stores acc[0].
// ---------------------------------------------------------------------------
template <int N, int K>
__device__ __forceinline__ void matvec_wmma(const unsigned short* __restrict__ Wp,
                                            const float* __restrict__ bias,
                                            float bscale,
                                            const __bf16* __restrict__ xb,
                                            float* __restrict__ y, int tid) {
    constexpr int Ntiles = N >> 4;
    constexpr int Ktiles = K >> 5;
    const int lane  = tid & 31;
    const int wave  = tid >> 5;
    const int abase = (lane < 16) ? 0 : 16;

    if constexpr (K == 256) {
        v16bf af[Ktiles];
#pragma unroll
        for (int kt = 0; kt < Ktiles; ++kt) af[kt] = a_frag(xb, kt, abase);
        for (int nt = wave; nt < Ntiles; nt += NW) {
            float bv = bias[nt * 16 + (lane & 15)] * bscale;
            v8f acc = {bv, bv, bv, bv, bv, bv, bv, bv};
            const unsigned short* wrow =
                Wp + (size_t)nt * Ktiles * 512 + (size_t)lane * 16;
#pragma unroll
            for (int kt = 0; kt < Ktiles; ++kt) {
                v16bf b = *(const v16bf*)(wrow + (size_t)kt * 512);
                acc = wmma_bf16(af[kt], b, acc);
            }
            if (lane < 16) y[nt * 16 + lane] = acc[0];
        }
    } else {
        for (int nt = wave; nt < Ntiles; nt += NW) {
            float bv = bias[nt * 16 + (lane & 15)] * bscale;
            v8f acc = {bv, bv, bv, bv, bv, bv, bv, bv};
            const unsigned short* wrow =
                Wp + (size_t)nt * Ktiles * 512 + (size_t)lane * 16;
#pragma unroll 8
            for (int kt = 0; kt < Ktiles; ++kt) {
                v16bf a = a_frag(xb, kt, abase);
                v16bf b = *(const v16bf*)(wrow + (size_t)kt * 512);
                acc = wmma_bf16(a, b, acc);
            }
            if (lane < 16) y[nt * 16 + lane] = acc[0];
        }
    }
}

__device__ __forceinline__ float wave_sum(float v) {
#pragma unroll
    for (int off = 16; off > 0; off >>= 1) v += __shfl_xor(v, off, 32);
    return v;
}
__device__ __forceinline__ float wave_max(float v) {
#pragma unroll
    for (int off = 16; off > 0; off >>= 1) v = fmaxf(v, __shfl_xor(v, off, 32));
    return v;
}

// Block moments over 512 threads (inactive threads contribute r = 0);
// single barrier window.  red must hold 32 floats.
struct Moments { float mean, rstd; };
__device__ __forceinline__ Moments block_moments(float r, int tid, float* red) {
    float s1 = wave_sum(r);
    float s2 = wave_sum(r * r);
    int w = tid >> 5;
    if ((tid & 31) == 0) { red[w] = s1; red[NW + w] = s2; }
    __syncthreads();
    float sum = 0.f, sq = 0.f;
#pragma unroll
    for (int k = 0; k < NW; ++k) { sum += red[k]; sq += red[NW + k]; }
    Moments m;
    m.mean = sum * (1.0f / D);
    float var = sq * (1.0f / D) - m.mean * m.mean;
    m.rstd = rsqrtf(var + EPSLN);
    return m;
}

__global__ __launch_bounds__(TPB) void decode_kernel(
    const float* __restrict__ x0,   const float* __restrict__ emb_w,
    const float* __restrict__ emb_b,
    const float* __restrict__ bq,   const float* __restrict__ bk,
    const float* __restrict__ bv,   const float* __restrict__ bo,
    const float* __restrict__ ln1w, const float* __restrict__ ln1b,
    const float* __restrict__ b1,   const float* __restrict__ b2,
    const float* __restrict__ ln2w, const float* __restrict__ ln2b,
    const float* __restrict__ lnfw, const float* __restrict__ lnfb,
    const float* __restrict__ outw, const float* __restrict__ outb,
    const unsigned short* __restrict__ WqP, const unsigned short* __restrict__ WkP,
    const unsigned short* __restrict__ WvP, const unsigned short* __restrict__ WoP,
    const unsigned short* __restrict__ W1P, const unsigned short* __restrict__ W2P,
    const float* __restrict__ pe, float* __restrict__ Kc, float* __restrict__ Vc,
    float* __restrict__ out) {

    __shared__ float traj[S * 32];
    __shared__ float xbuf[D];
    __shared__ float tmp[FF];
    __shared__ float qbuf[D];
    __shared__ float ksh[D];
    __shared__ float vsh[D];
    __shared__ float obuf[D];
    __shared__ float lg[NH * S];
    __shared__ float red[2 * NW];
    __shared__ alignas(32) __bf16 bfa[FF];   // bf16 matvec-input mirror

    const int tid  = threadIdx.x;
    const int lane = tid & 31;
    const int wave = tid >> 5;

    for (int idx = tid; idx < S * 32; idx += TPB) traj[idx] = 0.f;
    __syncthreads();
    if (tid < 32) { traj[tid] = x0[tid]; out[tid] = x0[tid]; }
    __syncthreads();

    const float qscale = 0.17677669529663687f;   // 1/sqrt(32)

    for (int i = 0; i < S; ++i) {
        // ---- embedding of token i: (emb_w @ traj_i + emb_b)*sqrt(d) + pe[i] ----
        if (tid < D) {
            float dot = 0.f;
            const float* er = emb_w + tid * 32;
#pragma unroll
            for (int k2 = 0; k2 < 32; ++k2) dot += er[k2] * traj[i * 32 + k2];
            float v = (dot + emb_b[tid]) * 16.0f + pe[i * D + tid];
            xbuf[tid] = v;
            bfa[tid]  = (__bf16)v;
        }
        __syncthreads();

        for (int l = 0; l < NL; ++l) {
            const unsigned short* Wql = WqP + (size_t)l * D * D;
            const unsigned short* Wkl = WkP + (size_t)l * D * D;
            const unsigned short* Wvl = WvP + (size_t)l * D * D;
            const unsigned short* Wol = WoP + (size_t)l * D * D;
            const unsigned short* W1l = W1P + (size_t)l * FF * D;
            const unsigned short* W2l = W2P + (size_t)l * D * FF;
            float* Kcl = Kc + (size_t)l * S * D;
            float* Vcl = Vc + (size_t)l * S * D;

            // ---- q, k, v projections, biases fused; one barrier total ----
            matvec_wmma<D, D>(Wql, bq + l * D, qscale, bfa, qbuf, tid);
            matvec_wmma<D, D>(Wkl, bk + l * D, 1.0f,   bfa, ksh,  tid);
            matvec_wmma<D, D>(Wvl, bv + l * D, 1.0f,   bfa, vsh,  tid);
            __syncthreads();

            // ---- KV-cache append (used by future steps / later stages) + logits ----
            if (tid < D) {
                Kcl[(size_t)i * D + tid] = ksh[tid];
                Vcl[(size_t)i * D + tid] = vsh[tid];
            }
            for (int hp = tid; hp < NH * S; hp += TPB) {
                int h = hp >> 7;          // /128
                int p = hp & 127;
                if (p <= i) {
                    const float* kr = (p == i) ? (ksh + h * HD)
                                               : (Kcl + (size_t)p * D + h * HD);
                    const float* qr = qbuf + h * HD;
                    float s = 0.f;
#pragma unroll
                    for (int d2 = 0; d2 < HD; ++d2) s += qr[d2] * kr[d2];
                    lg[hp] = s;
                }
            }
            __syncthreads();

            // ---- softmax: wave h owns head h (waves 8..15 idle here) ----
            if (wave < NH) {
                float mx = -1e30f;
                for (int p = lane; p <= i; p += 32) mx = fmaxf(mx, lg[wave * S + p]);
                mx = wave_max(mx);
                float sum = 0.f;
                for (int p = lane; p <= i; p += 32) {
                    float e = __expf(lg[wave * S + p] - mx);
                    lg[wave * S + p] = e;
                    sum += e;
                }
                sum = wave_sum(sum);
                float inv = 1.0f / sum;
                for (int p = lane; p <= i; p += 32) lg[wave * S + p] *= inv;
            }
            __syncthreads();

            // ---- weighted sum of V, split over 2 position groups ----
            {
                int sub = tid >> 8;       // 0 / 1
                int d   = tid & 255;
                int h   = d >> 5;         // HD == 32
                const float* aw = lg + h * S;
                float acc = 0.f;
                for (int p = sub; p <= i; p += 2)
                    acc += aw[p] * Vcl[(size_t)p * D + d];   // pos i read from L0-resident cache
                tmp[tid] = acc;
            }
            __syncthreads();
            if (tid < D) bfa[tid] = (__bf16)(tmp[tid] + tmp[D + tid]);
            __syncthreads();

            // ---- output projection (bo fused) + residual + LN1 ----
            matvec_wmma<D, D>(Wol, bo + l * D, 1.0f, bfa, tmp, tid);
            __syncthreads();
            {
                float r = 0.f;
                if (tid < D) r = xbuf[tid] + tmp[tid];
                Moments mo = block_moments(r, tid, red);
                if (tid < D) {
                    float nrm = (r - mo.mean) * mo.rstd * ln1w[l * D + tid] + ln1b[l * D + tid];
                    xbuf[tid] = nrm;
                    bfa[tid]  = (__bf16)nrm;
                }
            }
            __syncthreads();

            // ---- FFN: gelu(W1 x + b1), then W2 (b1, b2 fused) ----
            matvec_wmma<FF, D>(W1l, b1 + l * FF, 1.0f, bfa, tmp, tid);
            __syncthreads();
            for (int f = tid; f < FF; f += TPB) {
                float hh = tmp[f];
                float c  = 0.7978845608028654f * (hh + 0.044715f * hh * hh * hh);
                float g  = 0.5f * hh * (1.0f + tanhf(c));
                bfa[f]   = (__bf16)g;
            }
            __syncthreads();
            matvec_wmma<D, FF>(W2l, b2 + l * D, 1.0f, bfa, tmp, tid);
            __syncthreads();
            {
                float r = 0.f;
                if (tid < D) r = xbuf[tid] + tmp[tid];
                Moments mo = block_moments(r, tid, red);
                if (tid < D) {
                    float nrm = (r - mo.mean) * mo.rstd * ln2w[l * D + tid] + ln2b[l * D + tid];
                    xbuf[tid] = nrm;
                    bfa[tid]  = (__bf16)nrm;
                }
            }
            __syncthreads();
        }

        // ---- final LN ----
        {
            float r = (tid < D) ? xbuf[tid] : 0.f;
            Moments mo = block_moments(r, tid, red);
            if (tid < D)
                obuf[tid] = (r - mo.mean) * mo.rstd * lnfw[tid] + lnfb[tid];
        }
        __syncthreads();

        // ---- delta = out_w @ h + out_b ; 16 threads per output row ----
        {
            int m   = tid >> 4;           // 0..31
            int sub = tid & 15;
            const float* wr = outw + (size_t)m * D;
            float part = 0.f;
            for (int d2 = sub; d2 < D; d2 += 16) part += wr[d2] * obuf[d2];
#pragma unroll
            for (int off = 8; off > 0; off >>= 1) part += __shfl_xor(part, off, 32);
            if (sub == 0) {
                float nxt = traj[i * 32 + m] + part + outb[m];
                int j = (i + 1 < S) ? (i + 1) : (S - 1);
                traj[j * 32 + m] = nxt;
                out[j * 32 + m]  = nxt;
            }
        }
        __syncthreads();
    }
}

// ---------------------------------------------------------------------------
extern "C" void kernel_launch(void* const* d_in, const int* in_sizes, int n_in,
                              void* d_out, int out_size, void* d_ws, size_t ws_size,
                              hipStream_t stream) {
    (void)in_sizes; (void)n_in; (void)out_size; (void)ws_size;
    const float* x0    = (const float*)d_in[0];
    const float* emb_w = (const float*)d_in[1];
    const float* emb_b = (const float*)d_in[2];
    const float* Wq    = (const float*)d_in[3];
    const float* bq    = (const float*)d_in[4];
    const float* Wk    = (const float*)d_in[5];
    const float* bk    = (const float*)d_in[6];
    const float* Wv    = (const float*)d_in[7];
    const float* bv    = (const float*)d_in[8];
    const float* Wo    = (const float*)d_in[9];
    const float* bo    = (const float*)d_in[10];
    const float* ln1w  = (const float*)d_in[11];
    const float* ln1b  = (const float*)d_in[12];
    const float* W1    = (const float*)d_in[13];
    const float* b1    = (const float*)d_in[14];
    const float* W2    = (const float*)d_in[15];
    const float* b2    = (const float*)d_in[16];
    const float* ln2w  = (const float*)d_in[17];
    const float* ln2b  = (const float*)d_in[18];
    const float* lnfw  = (const float*)d_in[19];
    const float* lnfb  = (const float*)d_in[20];
    const float* outw  = (const float*)d_in[21];
    const float* outb  = (const float*)d_in[22];
    float* out = (float*)d_out;

    char* ws = (char*)d_ws;
    size_t off = 0;
    auto take = [&](size_t bytes) -> void* {
        void* p = ws + off;
        off += (bytes + 255) & ~(size_t)255;
        return p;
    };
    unsigned short* WqP = (unsigned short*)take((size_t)NL * D * D * 2);
    unsigned short* WkP = (unsigned short*)take((size_t)NL * D * D * 2);
    unsigned short* WvP = (unsigned short*)take((size_t)NL * D * D * 2);
    unsigned short* WoP = (unsigned short*)take((size_t)NL * D * D * 2);
    unsigned short* W1P = (unsigned short*)take((size_t)NL * FF * D * 2);
    unsigned short* W2P = (unsigned short*)take((size_t)NL * D * FF * 2);
    float* peT = (float*)take((size_t)S * D * 4);
    float* Kc  = (float*)take((size_t)NL * S * D * 4);
    float* Vc  = (float*)take((size_t)NL * S * D * 4);

    const float qscale = 0.17677669529663687f;   // 1/sqrt(32)
    for (int l = 0; l < NL; ++l) {
        pack_bf16_wmma<<<128, 256, 0, stream>>>(Wq + (size_t)l * D * D,  WqP + (size_t)l * D * D,  D, D, qscale);
        pack_bf16_wmma<<<128, 256, 0, stream>>>(Wk + (size_t)l * D * D,  WkP + (size_t)l * D * D,  D, D, 1.0f);
        pack_bf16_wmma<<<128, 256, 0, stream>>>(Wv + (size_t)l * D * D,  WvP + (size_t)l * D * D,  D, D, 1.0f);
        pack_bf16_wmma<<<128, 256, 0, stream>>>(Wo + (size_t)l * D * D,  WoP + (size_t)l * D * D,  D, D, 1.0f);
        pack_bf16_wmma<<<256, 256, 0, stream>>>(W1 + (size_t)l * FF * D, W1P + (size_t)l * FF * D, FF, D, 1.0f);
        pack_bf16_wmma<<<256, 256, 0, stream>>>(W2 + (size_t)l * D * FF, W2P + (size_t)l * D * FF, D, FF, 1.0f);
    }
    pe_kernel<<<(S * D + 255) / 256, 256, 0, stream>>>(peT);

    decode_kernel<<<1, TPB, 0, stream>>>(
        x0, emb_w, emb_b, bq, bk, bv, bo, ln1w, ln1b, b1, b2, ln2w, ln2b,
        lnfw, lnfb, outw, outb, WqP, WkP, WvP, WoP, W1P, W2P, peT, Kc, Vc, out);
}
